// DynamicsDecoder_90933047591474
// MI455X (gfx1250) — compile-verified, matrix operand order checked by reading, not run on hardware
//
#include <hip/hip_runtime.h>
#include <hip/hip_bf16.h>

#define HID 256
#define HEADS 4
#define IN_DIM 128
#define LAYERS 3

typedef __attribute__((ext_vector_type(16))) __bf16 v16bf;
typedef __attribute__((ext_vector_type(8)))  __bf16 v8bf;
typedef __attribute__((ext_vector_type(8)))  float  v8f;

// LDS B-tile: 256 columns, column stride 40 bf16 (80 bytes, 16B aligned, bank-spread)
#define BCOL_STRIDE 40

// ---------- helpers: order-preserving float<->uint for atomic max ----------
__device__ __forceinline__ unsigned f2o(float f) {
    unsigned b = __float_as_uint(f);
    return (b & 0x80000000u) ? ~b : (b | 0x80000000u);
}
__device__ __forceinline__ float o2f(unsigned u) {
    unsigned b = (u & 0x80000000u) ? (u & 0x7FFFFFFFu) : ~u;
    return __uint_as_float(b);
}

// A fragment (16-bit A layout): elements 0..7 = K kh..kh+7, 8..15 = K kh+16..kh+23
__device__ __forceinline__ v16bf load_a_frag(const float* __restrict__ p) {
    const float4 q0 = *(const float4*)(p);
    const float4 q1 = *(const float4*)(p + 4);
    const float4 q2 = *(const float4*)(p + 16);
    const float4 q3 = *(const float4*)(p + 20);
    v16bf a;
    a[0]  = (__bf16)q0.x; a[1]  = (__bf16)q0.y; a[2]  = (__bf16)q0.z; a[3]  = (__bf16)q0.w;
    a[4]  = (__bf16)q1.x; a[5]  = (__bf16)q1.y; a[6]  = (__bf16)q1.z; a[7]  = (__bf16)q1.w;
    a[8]  = (__bf16)q2.x; a[9]  = (__bf16)q2.y; a[10] = (__bf16)q2.z; a[11] = (__bf16)q2.w;
    a[12] = (__bf16)q3.x; a[13] = (__bf16)q3.y; a[14] = (__bf16)q3.z; a[15] = (__bf16)q3.w;
    return a;
}

// ---------------------------------------------------------------------------
// Tiled GEMM: out[N,256] = act(A[N,K] @ W[K,256] + bias)
// bf16 WMMA, fp32 accumulate. Block = 256 threads = 8 waves.
// Block tile: 64 rows x 256 cols. Wave tile: 32 rows x 64 cols = 8 WMMAs/K-step.
// Weight strip (32 x 256) staged in LDS as bf16 (double-buffered). All four B
// fragments are loaded into distinct registers up front (8x ds_load_b128 in
// one clause) so the WMMA pipe isn't stalled on s_wait_dscnt==0 per tile.
// ---------------------------------------------------------------------------
__global__ void __launch_bounds__(256)
gemm_bf16_wmma_kernel(const float* __restrict__ A,
                      const float* __restrict__ W,
                      const float* __restrict__ bias,
                      float* __restrict__ out,
                      int N, int K, int do_relu)
{
    __shared__ __attribute__((aligned(16))) unsigned short ldsB[2][256 * BCOL_STRIDE];

    const int tid    = threadIdx.x;
    const int lane   = tid & 31;
    const int wv     = tid >> 5;
    const int rowsub = wv & 1;          // 0: rows 0-31, 1: rows 32-63 of block tile
    const int colgrp = wv >> 1;         // 0..3 -> 64-col group
    const int m      = lane & 15;
    const int kh     = (lane >> 4) * 8; // 0 or 8
    const int col0   = colgrp * 64;
    const int rbase  = blockIdx.x * 64 + rowsub * 32;

    int r0 = rbase + m;      if (r0 > N - 1) r0 = N - 1;   // clamp, stores guarded
    int r1 = rbase + 16 + m; if (r1 > N - 1) r1 = N - 1;
    const float* __restrict__ Ar0 = A + (size_t)r0 * K;
    const float* __restrict__ Ar1 = A + (size_t)r1 * K;

    // each thread stages one column of the 32 x 256 weight strip
    const float* __restrict__ Wcol = W + tid;
    float wreg[32];

    v8f acc[2][4] = {};

    const int ksteps = K >> 5;

    // ---- stage first strip ----
    #pragma unroll
    for (int kk = 0; kk < 32; ++kk) wreg[kk] = Wcol[(size_t)kk * HID];
    {
        unsigned* dst = (unsigned*)(&ldsB[0][(size_t)tid * BCOL_STRIDE]);
        #pragma unroll
        for (int kk = 0; kk < 16; ++kk) {
            union { __bf16 h[2]; unsigned u; } t;
            t.h[0] = (__bf16)wreg[2 * kk];
            t.h[1] = (__bf16)wreg[2 * kk + 1];
            dst[kk] = t.u;
        }
    }
    __syncthreads();

    int buf = 0;
    for (int kt = 0; kt < ksteps; ++kt) {
        const int k0 = kt * 32;

        // prefetch next weight strip into registers (overlaps WMMA issue)
        if (kt + 1 < ksteps) {
            #pragma unroll
            for (int kk = 0; kk < 32; ++kk)
                wreg[kk] = Wcol[(size_t)(k0 + 32 + kk) * HID];
        }

        const v16bf a0 = load_a_frag(Ar0 + k0 + kh);
        const v16bf a1 = load_a_frag(Ar1 + k0 + kh);

        // hoist all B-fragment LDS loads into distinct registers (one clause)
        const unsigned short* __restrict__ bb = ldsB[buf];
        v16bf bfr[4];
        #pragma unroll
        for (int t = 0; t < 4; ++t) {
            const int c = col0 + t * 16 + m;
            const unsigned short* colp = bb + (size_t)c * BCOL_STRIDE;
            const v8bf blo = *(const v8bf*)(colp + kh);        // K kh..kh+7
            const v8bf bhi = *(const v8bf*)(colp + 16 + kh);   // K 16+kh..16+kh+7
            bfr[t] = __builtin_shufflevector(blo, bhi,
                0, 1, 2, 3, 4, 5, 6, 7, 8, 9, 10, 11, 12, 13, 14, 15);
        }

        // back-to-back WMMA issue
        #pragma unroll
        for (int t = 0; t < 4; ++t) {
            acc[0][t] = __builtin_amdgcn_wmma_f32_16x16x32_bf16(
                false, a0, false, bfr[t], (short)0, acc[0][t], false, false);
            acc[1][t] = __builtin_amdgcn_wmma_f32_16x16x32_bf16(
                false, a1, false, bfr[t], (short)0, acc[1][t], false, false);
        }

        // write next strip into the other buffer
        if (kt + 1 < ksteps) {
            unsigned* dst = (unsigned*)(&ldsB[buf ^ 1][(size_t)tid * BCOL_STRIDE]);
            #pragma unroll
            for (int kk = 0; kk < 16; ++kk) {
                union { __bf16 h[2]; unsigned u; } t;
                t.h[0] = (__bf16)wreg[2 * kk];
                t.h[1] = (__bf16)wreg[2 * kk + 1];
                dst[kk] = t.u;
            }
        }
        __syncthreads();
        buf ^= 1;
    }

    // ---- epilogue: C/D layout -> row = base + i + 8*(lane/16), col = lane%16
    #pragma unroll
    for (int rs = 0; rs < 2; ++rs) {
        #pragma unroll
        for (int t = 0; t < 4; ++t) {
            const int col  = col0 + t * 16 + m;
            const float bv = bias ? bias[col] : 0.0f;
            #pragma unroll
            for (int i = 0; i < 8; ++i) {
                const int r = rbase + rs * 16 + kh + i;
                if (r < N) {
                    float v = acc[rs][t][i] + bv;
                    if (do_relu) v = fmaxf(v, 0.0f);
                    out[(size_t)r * HID + col] = v;
                }
            }
        }
    }
}

// ---------------------------------------------------------------------------
__global__ void fill_u32_kernel(unsigned* __restrict__ p, unsigned v, int n)
{
    int i = blockIdx.x * blockDim.x + threadIdx.x;
    if (i < n) p[i] = v;
}

// ---------------------------------------------------------------------------
// Pass 1: per-edge logits + segment max (wave per edge)
// ---------------------------------------------------------------------------
__global__ void edge_logits_kernel(const float* __restrict__ xl,
                                   const float* __restrict__ xr,
                                   const long long* __restrict__ ei,
                                   const float* __restrict__ att,
                                   float* __restrict__ logits,
                                   unsigned* __restrict__ segmax,
                                   int E, int EN)
{
    const int lane = threadIdx.x & 31;
    const int edge = blockIdx.x * 8 + (threadIdx.x >> 5);
    if (edge >= EN) return;

    long long s, d;
    if (edge < E) { s = ei[edge]; d = ei[(size_t)E + edge]; }
    else          { s = d = (long long)(edge - E); }

    const float* __restrict__ pl = xl + (size_t)s * HID;
    const float* __restrict__ pr = xr + (size_t)d * HID;

    float part[HEADS] = {0.f, 0.f, 0.f, 0.f};
    #pragma unroll
    for (int j = 0; j < 8; ++j) {
        const int c = lane + 32 * j;          // head = c>>6 == j>>1 (lane-invariant)
        float v = pl[c] + pr[c];
        v = (v > 0.f) ? v : 0.2f * v;         // leaky_relu(0.2)
        part[j >> 1] += v * att[c];
    }

    #pragma unroll
    for (int hh = 0; hh < HEADS; ++hh) {
        float v = part[hh];
        #pragma unroll
        for (int off = 16; off > 0; off >>= 1) v += __shfl_xor(v, off);
        if (lane == 0) {
            logits[(size_t)edge * HEADS + hh] = v;
            atomicMax(&segmax[(size_t)d * HEADS + hh], f2o(v));
        }
    }
}

// ---------------------------------------------------------------------------
// Pass 2: a = exp(logit - max[dst]); segment sum (thread per edge*head)
// ---------------------------------------------------------------------------
__global__ void edge_softmax_kernel(float* __restrict__ logits,
                                    const unsigned* __restrict__ segmax,
                                    float* __restrict__ segsum,
                                    const long long* __restrict__ ei,
                                    int E, int EN)
{
    const int idx = blockIdx.x * blockDim.x + threadIdx.x;
    if (idx >= EN * HEADS) return;
    const int edge = idx >> 2;
    const int hh   = idx & 3;

    long long d = (edge < E) ? ei[(size_t)E + edge] : (long long)(edge - E);
    const float mv = o2f(segmax[(size_t)d * HEADS + hh]);
    const float a  = expf(logits[idx] - mv);
    logits[idx] = a;
    atomicAdd(&segsum[(size_t)d * HEADS + hh], a);
}

// ---------------------------------------------------------------------------
// Pass 3: agg[dst] += (a / max(sum[dst],1e-16)) * xl[src]  (wave per edge)
// ---------------------------------------------------------------------------
__global__ void edge_aggregate_kernel(const float* __restrict__ xl,
                                      const float* __restrict__ alpha,
                                      const float* __restrict__ segsum,
                                      const long long* __restrict__ ei,
                                      float* __restrict__ agg,
                                      int E, int EN)
{
    const int lane = threadIdx.x & 31;
    const int edge = blockIdx.x * 8 + (threadIdx.x >> 5);
    if (edge >= EN) return;

    long long s, d;
    if (edge < E) { s = ei[edge]; d = ei[(size_t)E + edge]; }
    else          { s = d = (long long)(edge - E); }

    float al[HEADS];
    #pragma unroll
    for (int hh = 0; hh < HEADS; ++hh) {
        const float ssum = segsum[(size_t)d * HEADS + hh];
        al[hh] = alpha[(size_t)edge * HEADS + hh] / fmaxf(ssum, 1e-16f);
    }

    const float* __restrict__ pl = xl + (size_t)s * HID;
    float* __restrict__ pd = agg + (size_t)d * HID;
    #pragma unroll
    for (int j = 0; j < 8; ++j) {
        const int c = lane + 32 * j;
        atomicAdd(&pd[c], al[j >> 1] * pl[c]);
    }
}

// ---------------------------------------------------------------------------
// h = relu(layernorm(agg + gat_b + skip))  (wave per row)
// ---------------------------------------------------------------------------
__global__ void ln_relu_kernel(const float* __restrict__ agg,
                               const float* __restrict__ gat_b,
                               const float* __restrict__ skip,
                               const float* __restrict__ g,
                               const float* __restrict__ b,
                               float* __restrict__ out, int N)
{
    const int lane = threadIdx.x & 31;
    const int row  = blockIdx.x * 8 + (threadIdx.x >> 5);
    if (row >= N) return;

    float v[8];
    float s = 0.f;
    #pragma unroll
    for (int j = 0; j < 8; ++j) {
        const int c = lane + 32 * j;
        v[j] = agg[(size_t)row * HID + c] + gat_b[c] + skip[(size_t)row * HID + c];
        s += v[j];
    }
    #pragma unroll
    for (int off = 16; off > 0; off >>= 1) s += __shfl_xor(s, off);
    const float mu = s * (1.0f / HID);

    float var = 0.f;
    #pragma unroll
    for (int j = 0; j < 8; ++j) { const float dl = v[j] - mu; var += dl * dl; }
    #pragma unroll
    for (int off = 16; off > 0; off >>= 1) var += __shfl_xor(var, off);
    const float rstd = rsqrtf(var * (1.0f / HID) + 1e-5f);

    #pragma unroll
    for (int j = 0; j < 8; ++j) {
        const int c = lane + 32 * j;
        const float o = g[c] * (v[j] - mu) * rstd + b[c];
        out[(size_t)row * HID + c] = fmaxf(o, 0.0f);
    }
}

// ---------------------------------------------------------------------------
// out[n] = t[n,:] @ W2[:,0] + b2  (wave per row)
// ---------------------------------------------------------------------------
__global__ void out_head_kernel(const float* __restrict__ t,
                                const float* __restrict__ W2,
                                const float* __restrict__ b2,
                                float* __restrict__ out, int N)
{
    const int lane = threadIdx.x & 31;
    const int row  = blockIdx.x * 8 + (threadIdx.x >> 5);
    if (row >= N) return;

    float s = 0.f;
    #pragma unroll
    for (int j = 0; j < 8; ++j) {
        const int c = lane + 32 * j;
        s += t[(size_t)row * HID + c] * W2[c];
    }
    #pragma unroll
    for (int off = 16; off > 0; off >>= 1) s += __shfl_xor(s, off);
    if (lane == 0) out[row] = s + b2[0];
}

// ---------------------------------------------------------------------------
extern "C" void kernel_launch(void* const* d_in, const int* in_sizes, int n_in,
                              void* d_out, int out_size, void* d_ws, size_t ws_size,
                              hipStream_t stream)
{
    (void)n_in; (void)out_size; (void)ws_size;

    const float*     x      = (const float*)d_in[0];
    const long long* ei     = (const long long*)d_in[1];
    // d_in[2] batch, d_in[3] adj_matrices: unused by the reference math
    const float*     enc_W  = (const float*)d_in[4];
    const float*     enc_b  = (const float*)d_in[5];
    const float*     Wl     = (const float*)d_in[6];
    const float*     Wr     = (const float*)d_in[7];
    const float*     att    = (const float*)d_in[8];
    const float*     gat_b  = (const float*)d_in[9];
    const float*     skip_W = (const float*)d_in[10];
    const float*     skip_b = (const float*)d_in[11];
    const float*     ln_g   = (const float*)d_in[12];
    const float*     ln_b   = (const float*)d_in[13];
    const float*     out_W1 = (const float*)d_in[14];
    const float*     out_b1 = (const float*)d_in[15];
    const float*     out_W2 = (const float*)d_in[16];
    const float*     out_b2 = (const float*)d_in[17];

    const int N  = in_sizes[0] / IN_DIM;   // 25000
    const int E  = in_sizes[1] / 2;        // 500000
    const int EN = E + N;                  // edges + self loops

    // ---- workspace layout (floats) ----
    float* ws = (float*)d_ws;
    const size_t nh = (size_t)N * HID;
    float*    h      = ws;                                      // [N,256]
    float*    xl     = h + nh;                                  // [N,256]
    float*    xr     = xl + nh;                                 // [N,256] (reused as skip out)
    float*    agg    = xr + nh;                                 // [N,256]
    float*    logits = agg + nh;                                // [EN,4] -> alpha in place
    float*    segsum = logits + (size_t)EN * HEADS;             // [N,4]
    unsigned* segmax = (unsigned*)(segsum + (size_t)N * HEADS); // [N,4]

    const int gemmGrid = (N + 63) / 64;
    const dim3 gemmBlk(256);

    // encoder: h = relu(x @ enc_W + enc_b), K = 128
    gemm_bf16_wmma_kernel<<<gemmGrid, gemmBlk, 0, stream>>>(x, enc_W, enc_b, h, N, IN_DIM, 1);

    const int edgeBlocks = (EN + 7) / 8;
    const int ehBlocks   = (EN * HEADS + 255) / 256;
    const int rowBlocks  = (N + 7) / 8;
    const int fillSeg    = (N * HEADS + 255) / 256;
    const int fillAgg    = ((int)nh + 255) / 256;

    for (int l = 0; l < LAYERS; ++l) {
        const float* Wl_l   = Wl     + (size_t)l * HID * HID;
        const float* Wr_l   = Wr     + (size_t)l * HID * HID;
        const float* att_l  = att    + (size_t)l * HID;       // 4*64 = 256
        const float* gatb_l = gat_b  + (size_t)l * HID;
        const float* skW_l  = skip_W + (size_t)l * HID * HID;
        const float* skb_l  = skip_b + (size_t)l * HID;
        const float* lng_l  = ln_g   + (size_t)l * HID;
        const float* lnb_l  = ln_b   + (size_t)l * HID;

        // xl = h @ Wl, xr = h @ Wr (no bias, no act)
        gemm_bf16_wmma_kernel<<<gemmGrid, gemmBlk, 0, stream>>>(h, Wl_l, nullptr, xl, N, HID, 0);
        gemm_bf16_wmma_kernel<<<gemmGrid, gemmBlk, 0, stream>>>(h, Wr_l, nullptr, xr, N, HID, 0);

        // init segment buffers: segmax = ord(-inf) = 0x007FFFFF, segsum = 0, agg = 0
        fill_u32_kernel<<<fillSeg, 256, 0, stream>>>(segmax, 0x007FFFFFu, N * HEADS);
        fill_u32_kernel<<<fillSeg, 256, 0, stream>>>((unsigned*)segsum, 0u, N * HEADS);
        fill_u32_kernel<<<fillAgg, 256, 0, stream>>>((unsigned*)agg, 0u, (int)nh);

        edge_logits_kernel<<<edgeBlocks, 256, 0, stream>>>(xl, xr, ei, att_l, logits, segmax, E, EN);
        edge_softmax_kernel<<<ehBlocks, 256, 0, stream>>>(logits, segmax, segsum, ei, E, EN);
        edge_aggregate_kernel<<<edgeBlocks, 256, 0, stream>>>(xl, logits, segsum, ei, agg, E, EN);

        // skip branch: xr <- relu(h @ skip_W + skip_b)
        gemm_bf16_wmma_kernel<<<gemmGrid, gemmBlk, 0, stream>>>(h, skW_l, skb_l, xr, N, HID, 1);

        // h <- relu(LN(agg + gat_b + skip))
        ln_relu_kernel<<<rowBlocks, 256, 0, stream>>>(agg, gatb_l, xr, lng_l, lnb_l, h, N);
    }

    // output head: xl <- relu(h @ out_W1 + out_b1); out <- xl @ out_W2 + out_b2
    gemm_bf16_wmma_kernel<<<gemmGrid, gemmBlk, 0, stream>>>(h, out_W1, out_b1, xl, N, HID, 1);
    out_head_kernel<<<rowBlocks, 256, 0, stream>>>(xl, out_W2, out_b2, (float*)d_out, N);
}
